// CAGECareRF_58488864637086
// MI455X (gfx1250) — matrix operand order repaired and checked by reading, not compiled, not run on hardware
//
#include <hip/hip_runtime.h>
#include <hip/hip_bf16.h>

#define N_NODES 50000
#define N_EDGES 800000
#define DIM     128
#define NP      50176      // 392 * 128, padded row count for full GEMM blocks
#define TOPK_K  10

typedef __bf16 bf16_t;
typedef __bf16 v4bf  __attribute__((ext_vector_type(4)));
typedef __bf16 v8bf  __attribute__((ext_vector_type(8)));
typedef __bf16 v16bf __attribute__((ext_vector_type(16)));
typedef float  v8f   __attribute__((ext_vector_type(8)));
typedef float  v4f   __attribute__((ext_vector_type(4)));

// ---------------- utility kernels ----------------

__global__ void k_fill_i32(int* __restrict__ p, int n, int v) {
  int t = blockIdx.x * blockDim.x + threadIdx.x;
  if (t < n) p[t] = v;
}

// vectorized f32 -> bf16 conversion (4 elems / thread)
__global__ void k_convert(const float* __restrict__ in, bf16_t* __restrict__ out, int n4) {
  int t = blockIdx.x * blockDim.x + threadIdx.x;
  if (t >= n4) return;
  v4f a = *(const v4f*)(in + (size_t)t * 4);
  v4bf o;
#pragma unroll
  for (int i = 0; i < 4; i++) o[i] = (bf16_t)a[i];
  *(v4bf*)(out + (size_t)t * 4) = o;
}

// pack a 128x128 fp32 row-major weight (W[k][n]) into bf16 B-operand fragment order:
// frag = kc*8+nt ; per-lane 16 contiguous bf16: lanes 0-15 hold K = kc*32 + 0..15,
// lanes 16-31 hold K = kc*32 + 16..31 ; column N = nt*16 + lane%16.
__global__ void k_pack(const float* __restrict__ W, bf16_t* __restrict__ out) {
  int t = blockIdx.x * blockDim.x + threadIdx.x;
  if (t >= 16384) return;
  int frag = t >> 9, lane = (t >> 4) & 31, j = t & 15;
  int kc = frag >> 3, nt = frag & 7;
  int K = kc * 32 + ((lane & 16) ? 16 : 0) + j;
  int Ncol = nt * 16 + (lane & 15);
  out[t] = (bf16_t)W[K * DIM + Ncol];
}

// ---------------- graph filtering (CARE top-k + Cheb normalization) ----------------

__global__ void k_invnorm(const float* __restrict__ x, float* __restrict__ invn) {
  int wid = (blockIdx.x * blockDim.x + threadIdx.x) >> 5;
  int lane = threadIdx.x & 31;
  if (wid >= N_NODES) return;
  v4f a = *(const v4f*)(x + (size_t)wid * DIM + lane * 4);
  float d = a[0]*a[0] + a[1]*a[1] + a[2]*a[2] + a[3]*a[3];
  for (int m = 16; m >= 1; m >>= 1) d += __shfl_xor(d, m, 32);
  if (lane == 0) invn[wid] = 1.0f / fmaxf(sqrtf(d), 1e-12f);
}

// one wave per edge: cosine sim + out-degree histogram
__global__ void k_edge_sim(const int* __restrict__ src, const int* __restrict__ dst,
                           const float* __restrict__ x, const float* __restrict__ invn,
                           float* __restrict__ sim, int* __restrict__ cnt) {
  int e = (blockIdx.x * blockDim.x + threadIdx.x) >> 5;
  int lane = threadIdx.x & 31;
  if (e >= N_EDGES) return;
  int s = src[e], d = dst[e];
  v4f a = *(const v4f*)(x + (size_t)s * DIM + lane * 4);
  v4f b = *(const v4f*)(x + (size_t)d * DIM + lane * 4);
  float dot = a[0]*b[0] + a[1]*b[1] + a[2]*b[2] + a[3]*b[3];
  for (int m = 16; m >= 1; m >>= 1) dot += __shfl_xor(dot, m, 32);
  if (lane == 0) {
    sim[e] = dot * invn[s] * invn[d];
    atomicAdd(&cnt[s], 1);
  }
}

// single-workgroup exclusive scan of n counts -> out[0..n], out[n] = total
__global__ void k_scan(const int* __restrict__ in, int* __restrict__ out, int n) {
  __shared__ int sm[1024];
  int t = threadIdx.x;
  int chunk = (n + 1023) >> 10;
  int s0 = t * chunk, s1 = s0 + chunk; if (s1 > n) s1 = n; if (s0 > n) s0 = n;
  int sum = 0;
  for (int i = s0; i < s1; i++) sum += in[i];
  sm[t] = sum;
  __syncthreads();
  for (int off = 1; off < 1024; off <<= 1) {
    int v = (t >= off) ? sm[t - off] : 0;
    __syncthreads();
    sm[t] += v;
    __syncthreads();
  }
  int run = sm[t] - sum;
  for (int i = s0; i < s1; i++) { out[i] = run; run += in[i]; }
  if (t == 1023) out[n] = sm[1023];
}

__global__ void k_scatter(const int* __restrict__ src, const int* __restrict__ dst,
                          const float* __restrict__ sim, const int* __restrict__ rowp,
                          int* __restrict__ cur, int* __restrict__ csr_dst,
                          float* __restrict__ csr_sim) {
  int e = blockIdx.x * blockDim.x + threadIdx.x;
  if (e >= N_EDGES) return;
  int s = src[e];
  int p = rowp[s] + atomicAdd(&cur[s], 1);
  csr_dst[p] = dst[e];
  csr_sim[p] = sim[e];
}

// per-node top-10 threshold selection + self-loop removal; kept out-degree + in-degree histogram
__global__ void k_topk(const int* __restrict__ rowp, const int* __restrict__ csr_dst,
                       const float* __restrict__ csr_sim, unsigned char* __restrict__ keep,
                       int* __restrict__ kdeg, int* __restrict__ indeg) {
  int n = blockIdx.x * blockDim.x + threadIdx.x;
  if (n >= N_NODES) return;
  int a = rowp[n], b = rowp[n + 1];
  int d = b - a;
  float thr = -3.4e38f;
  if (d > TOPK_K) {
    float top[TOPK_K];
    for (int i = 0; i < TOPK_K; i++) top[i] = -3.4e38f;
    for (int j = a; j < b; j++) {
      float s = csr_sim[j];
      if (s > top[TOPK_K - 1]) {
        int p = TOPK_K - 1;
        while (p > 0 && top[p - 1] < s) { top[p] = top[p - 1]; p--; }
        top[p] = s;
      }
    }
    thr = top[TOPK_K - 1];
  }
  int g = 0;
  for (int j = a; j < b; j++) if (csr_sim[j] > thr) g++;
  int quota = TOPK_K - g;
  int kept = 0;
  for (int j = a; j < b; j++) {
    float s = csr_sim[j];
    bool kf;
    if (s > thr) kf = true;
    else if (s == thr && quota > 0) { kf = true; quota--; }
    else kf = false;
    bool valid = kf && (csr_dst[j] != n);
    keep[j] = valid ? 1 : 0;
    if (valid) { kept++; atomicAdd(&indeg[csr_dst[j]], 1); }
  }
  kdeg[n] = kept;
}

__global__ void k_dinv(const int* __restrict__ kdeg, float* __restrict__ dinv) {
  int n = blockIdx.x * blockDim.x + threadIdx.x;
  if (n >= N_NODES) return;
  int k = kdeg[n];
  dinv[n] = (k > 0) ? rsqrtf((float)k) : 0.0f;
}

// build CSC (incoming) list of kept edges with weights w = -dinv[src]*dinv[dst]
__global__ void k_scatter_csc(const int* __restrict__ rowp, const int* __restrict__ csr_dst,
                              const unsigned char* __restrict__ keep,
                              const float* __restrict__ dinv, const int* __restrict__ colp,
                              int* __restrict__ cur2, int* __restrict__ csc_src,
                              float* __restrict__ csc_w) {
  int n = blockIdx.x * blockDim.x + threadIdx.x;
  if (n >= N_NODES) return;
  int a = rowp[n], b = rowp[n + 1];
  float di = dinv[n];
  for (int j = a; j < b; j++) {
    if (keep[j]) {
      int dn = csr_dst[j];
      float w = -di * dinv[dn];
      int p = colp[dn] + atomicAdd(&cur2[dn], 1);
      csc_src[p] = n;
      csc_w[p] = w;
    }
  }
}

// gather-form propagation: out[i] = sum_{(j->i) kept} w * h[j]   (atomic-free)
// prefetch two edges ahead to hide L2 latency of the random 512B row gathers
__global__ void k_prop(const int* __restrict__ colp, const int* __restrict__ csc_src,
                       const float* __restrict__ csc_w, const float* __restrict__ h,
                       float* __restrict__ out) {
  int wid = (blockIdx.x * blockDim.x + threadIdx.x) >> 5;
  int lane = threadIdx.x & 31;
  if (wid >= N_NODES) return;
  int a = colp[wid], b = colp[wid + 1];
  v4f acc = {0.f, 0.f, 0.f, 0.f};
  for (int j = a; j < b; j++) {
    if (j + 2 < b) {
      int sp = csc_src[j + 2];
      __builtin_prefetch(h + (size_t)sp * DIM + lane * 4, 0, 0);
    }
    int s = csc_src[j];
    float w = csc_w[j];
    v4f hv = *(const v4f*)(h + (size_t)s * DIM + lane * 4);
    acc += hv * w;
  }
  *(v4f*)(out + (size_t)wid * DIM + lane * 4) = acc;
}

// ---------------- dense GEMM via WMMA bf16 -> f32 ----------------
// D[NPx128] = A[NPx128]@B[128x128] (+C).  Block = 256 thr = 8 waves, 128 rows.
// B pre-packed in fragment order, staged through LDS, shared by all 8 waves.
// N-tiles processed in pairs with independent accumulators so dependent
// WMMA chains alternate (removes D->C serialization bubbles / v_nops).
__global__ void k_gemm(const bf16_t* __restrict__ A, const bf16_t* __restrict__ Bp,
                       const float* __restrict__ Cin, float* __restrict__ Dout) {
  __shared__ __align__(32) bf16_t Bsh[32 * 512];
  {
    const uint4* bsrc = (const uint4*)Bp;
    uint4* bdst = (uint4*)Bsh;
    for (int i = threadIdx.x; i < 2048; i += 256) bdst[i] = bsrc[i];
  }
  __syncthreads();
  int wave = threadIdx.x >> 5, lane = threadIdx.x & 31;
  int lm = lane & 15;
  bool hi = lane >= 16;
  int m0 = blockIdx.x * 128 + wave * 16;
  const bf16_t* arow = A + (size_t)(m0 + lm) * DIM;
  // warm L0 for the next block's A strip while we compute this one
  __builtin_prefetch(arow + (size_t)128 * DIM, 0, 0);
  // A fragments for the 4 K-chunks: two contiguous 8-elem runs per lane
  v16bf afrag[4];
#pragma unroll
  for (int kc = 0; kc < 4; kc++) {
    int base = kc * 32 + (hi ? 16 : 0);
    v8bf lo = *(const v8bf*)(arow + base);
    v8bf hh = *(const v8bf*)(arow + base + 8);
    v16bf a;
#pragma unroll
    for (int j = 0; j < 8; j++) { a[j] = lo[j]; a[j + 8] = hh[j]; }
    afrag[kc] = a;
  }
  int rbase = m0 + (hi ? 8 : 0);
#pragma unroll
  for (int ntp = 0; ntp < 4; ntp++) {
    int nt0 = ntp * 2, nt1 = nt0 + 1;
    int col0 = nt0 * 16 + lm, col1 = col0 + 16;
    v8f c0, c1;
    if (Cin) {
#pragma unroll
      for (int v = 0; v < 8; v++) {
        c0[v] = Cin[(size_t)(rbase + v) * DIM + col0];
        c1[v] = Cin[(size_t)(rbase + v) * DIM + col1];
      }
    } else {
#pragma unroll
      for (int v = 0; v < 8; v++) { c0[v] = 0.0f; c1[v] = 0.0f; }
    }
#pragma unroll
    for (int kc = 0; kc < 4; kc++) {
      v16bf b0 = *(const v16bf*)&Bsh[(kc * 8 + nt0) * 512 + lane * 16];
      v16bf b1 = *(const v16bf*)&Bsh[(kc * 8 + nt1) * 512 + lane * 16];
      c0 = __builtin_amdgcn_wmma_f32_16x16x32_bf16(false, afrag[kc], false, b0,
                                                   (short)0, c0, false, false);
      c1 = __builtin_amdgcn_wmma_f32_16x16x32_bf16(false, afrag[kc], false, b1,
                                                   (short)0, c1, false, false);
    }
#pragma unroll
    for (int v = 0; v < 8; v++) {
      Dout[(size_t)(rbase + v) * DIM + col0] = c0[v];
      Dout[(size_t)(rbase + v) * DIM + col1] = c1[v];
    }
  }
}

// out = relu(acc + bias [+ skip]); optional bf16 mirror for the next GEMM.
// 4 elems / thread (rows*32 threads)
__global__ void k_post(const float* __restrict__ acc, const float* __restrict__ bias,
                       const float* __restrict__ skip, float* __restrict__ out,
                       bf16_t* __restrict__ outb, int rows) {
  int t = blockIdx.x * blockDim.x + threadIdx.x;
  if (t >= rows * 32) return;
  int q = t & 31;
  size_t off = (size_t)t * 4;
  v4f v = *(const v4f*)(acc + off);
  v4f bv = *(const v4f*)(bias + q * 4);
  v += bv;
  if (skip) v += *(const v4f*)(skip + off);
#pragma unroll
  for (int i = 0; i < 4; i++) v[i] = fmaxf(v[i], 0.0f);
  *(v4f*)(out + off) = v;
  if (outb) {
    v4bf o;
#pragma unroll
    for (int i = 0; i < 4; i++) o[i] = (bf16_t)v[i];
    *(v4bf*)(outb + off) = o;
  }
}

// Chebyshev recurrence: out = 2*P - T0 (fp32 + bf16 mirror), 4 elems / thread
__global__ void k_combine(const float* __restrict__ P, const float* __restrict__ T0,
                          float* __restrict__ out, bf16_t* __restrict__ outb, int n4) {
  int t = blockIdx.x * blockDim.x + threadIdx.x;
  if (t >= n4) return;
  size_t off = (size_t)t * 4;
  v4f p = *(const v4f*)(P + off);
  v4f t0 = *(const v4f*)(T0 + off);
  v4f v = p * 2.0f - t0;
  *(v4f*)(out + off) = v;
  v4bf o;
#pragma unroll
  for (int i = 0; i < 4; i++) o[i] = (bf16_t)v[i];
  *(v4bf*)(outb + off) = o;
}

// out[i*stride] = dot(in[i,:], vec) + bias[0]   (one wave per row)
__global__ void k_matvec(const float* __restrict__ in, const float* __restrict__ vec,
                         const float* __restrict__ bias, float* __restrict__ out, int stride) {
  int wid = (blockIdx.x * blockDim.x + threadIdx.x) >> 5;
  int lane = threadIdx.x & 31;
  if (wid >= N_NODES) return;
  v4f a = *(const v4f*)(in + (size_t)wid * DIM + lane * 4);
  v4f v = *(const v4f*)(vec + lane * 4);
  float d = a[0]*v[0] + a[1]*v[1] + a[2]*v[2] + a[3]*v[3];
  for (int m = 16; m >= 1; m >>= 1) d += __shfl_xor(d, m, 32);
  if (lane == 0) out[(size_t)wid * stride] = d + bias[0];
}

// per-node softmax over 3 relation scores + weighted fuse (one wave per node)
__global__ void k_softmax_fuse(const float* __restrict__ score, const float* __restrict__ e0,
                               const float* __restrict__ e1, const float* __restrict__ e2,
                               float* __restrict__ fused) {
  int wid = (blockIdx.x * blockDim.x + threadIdx.x) >> 5;
  int lane = threadIdx.x & 31;
  if (wid >= N_NODES) return;
  float s0 = score[wid * 3 + 0], s1 = score[wid * 3 + 1], s2 = score[wid * 3 + 2];
  float m = fmaxf(s0, fmaxf(s1, s2));
  float x0 = expf(s0 - m), x1 = expf(s1 - m), x2 = expf(s2 - m);
  float inv = 1.0f / (x0 + x1 + x2);
  size_t off = (size_t)wid * DIM + lane * 4;
  v4f a = *(const v4f*)(e0 + off);
  v4f b = *(const v4f*)(e1 + off);
  v4f c = *(const v4f*)(e2 + off);
  v4f f = a * (x0 * inv) + b * (x1 * inv) + c * (x2 * inv);
  *(v4f*)(fused + off) = f;
}

// ---------------- driver ----------------

extern "C" void kernel_launch(void* const* d_in, const int* in_sizes, int n_in,
                              void* d_out, int out_size, void* d_ws, size_t ws_size,
                              hipStream_t stream) {
  const float* x      = (const float*)d_in[0];
  const int*   ei[3]  = {(const int*)d_in[1], (const int*)d_in[2], (const int*)d_in[3]};
  const float* conv_W = (const float*)d_in[4];
  const float* conv_b = (const float*)d_in[5];
  const float* gW1 = (const float*)d_in[6];
  const float* gb1 = (const float*)d_in[7];
  const float* gW2 = (const float*)d_in[8];
  const float* gb2 = (const float*)d_in[9];
  const float* pW  = (const float*)d_in[10];
  const float* pb  = (const float*)d_in[11];
  const float* cW1 = (const float*)d_in[12];
  const float* cb1 = (const float*)d_in[13];
  const float* cW2 = (const float*)d_in[14];
  const float* cb2 = (const float*)d_in[15];
  const float* auxW = (const float*)d_in[16];
  const float* auxb = (const float*)d_in[17];
  float* out = (float*)d_out;

  char* p = (char*)d_ws;
  auto alloc = [&](size_t bytes) -> void* {
    void* r = (void*)p;
    p += (bytes + 255) & ~(size_t)255;
    return r;
  };
  const size_t FB = (size_t)NP * DIM;
  float* emb[3];
  for (int r = 0; r < 3; r++) emb[r] = (float*)alloc(FB * 4);
  float* acc = (float*)alloc(FB * 4);
  float* P1  = (float*)alloc(FB * 4);
  float* P2  = (float*)alloc(FB * 4);
  bf16_t* XB  = (bf16_t*)alloc(FB * 2);
  bf16_t* HB  = (bf16_t*)alloc(FB * 2);
  bf16_t* T1B = (bf16_t*)alloc(FB * 2);
  bf16_t* T2B = (bf16_t*)alloc(FB * 2);
  bf16_t* packW = (bf16_t*)alloc((size_t)30 * 16384 * 2);
  float* sim      = (float*)alloc((size_t)N_EDGES * 4);
  int*   csr_dst  = (int*)alloc((size_t)N_EDGES * 4);
  float* csr_sim  = (float*)alloc((size_t)N_EDGES * 4);
  unsigned char* keep = (unsigned char*)alloc(N_EDGES);
  int*   csc_src  = (int*)alloc((size_t)N_EDGES * 4);
  float* csc_w    = (float*)alloc((size_t)N_EDGES * 4);
  int* cnt   = (int*)alloc((size_t)N_NODES * 4);
  int* rowp  = (int*)alloc((size_t)(N_NODES + 1) * 4);
  int* cur   = (int*)alloc((size_t)N_NODES * 4);
  int* kdeg  = (int*)alloc((size_t)N_NODES * 4);
  int* indeg = (int*)alloc((size_t)N_NODES * 4);
  int* colp  = (int*)alloc((size_t)(N_NODES + 1) * 4);
  int* cur2  = (int*)alloc((size_t)N_NODES * 4);
  float* dinv = (float*)alloc((size_t)N_NODES * 4);
  float* invn = (float*)alloc((size_t)N_NODES * 4);
  float* score = (float*)alloc((size_t)N_NODES * 3 * 4);

  dim3 blk(256);
  const int gN    = (N_NODES + 255) / 256;
  const int gNW   = (N_NODES * 32 + 255) / 256;   // 1 wave / node
  const int gE    = (N_EDGES + 255) / 256;
  const int gEW   = (N_EDGES * 32 + 255) / 256;   // 1 wave / edge
  const int gVec  = (N_NODES * 32 + 255) / 256;   // 4 elems / thread elementwise
  const int NV4   = N_NODES * 32;                 // number of float4 groups
  const int gGemm = NP / 128;                     // 392 full blocks

  // pack all 128x128 weights into bf16 B-fragment order
  for (int slot = 0; slot < 27; slot++)
    k_pack<<<64, blk, 0, stream>>>(conv_W + (size_t)slot * 16384, packW + (size_t)slot * 16384);
  k_pack<<<64, blk, 0, stream>>>(gW1, packW + (size_t)27 * 16384);
  k_pack<<<64, blk, 0, stream>>>(pW,  packW + (size_t)28 * 16384);
  k_pack<<<64, blk, 0, stream>>>(cW1, packW + (size_t)29 * 16384);

  k_invnorm<<<gNW, blk, 0, stream>>>(x, invn);
  k_convert<<<gVec, blk, 0, stream>>>(x, XB, NV4);

  for (int r = 0; r < 3; r++) {
    const int* src = ei[r];
    const int* dst = ei[r] + N_EDGES;
    k_fill_i32<<<gN, blk, 0, stream>>>(cnt, N_NODES, 0);
    k_edge_sim<<<gEW, blk, 0, stream>>>(src, dst, x, invn, sim, cnt);
    k_scan<<<1, 1024, 0, stream>>>(cnt, rowp, N_NODES);
    k_fill_i32<<<gN, blk, 0, stream>>>(cur, N_NODES, 0);
    k_scatter<<<gE, blk, 0, stream>>>(src, dst, sim, rowp, cur, csr_dst, csr_sim);
    k_fill_i32<<<gN, blk, 0, stream>>>(indeg, N_NODES, 0);
    k_topk<<<gN, blk, 0, stream>>>(rowp, csr_dst, csr_sim, keep, kdeg, indeg);
    k_dinv<<<gN, blk, 0, stream>>>(kdeg, dinv);
    k_scan<<<1, 1024, 0, stream>>>(indeg, colp, N_NODES);
    k_fill_i32<<<gN, blk, 0, stream>>>(cur2, N_NODES, 0);
    k_scatter_csc<<<gN, blk, 0, stream>>>(rowp, csr_dst, keep, dinv, colp, cur2, csc_src, csc_w);

    const float* inF = x;
    const bf16_t* inB = XB;
    float* h = emb[r];
    for (int l = 0; l < 3; l++) {
      int s0 = (r * 3 + l) * 3;
      k_gemm<<<gGemm, blk, 0, stream>>>(inB, packW + (size_t)(s0 + 0) * 16384, (const float*)nullptr, acc);
      k_prop<<<gNW, blk, 0, stream>>>(colp, csc_src, csc_w, inF, P1);
      k_convert<<<gVec, blk, 0, stream>>>(P1, T1B, NV4);
      k_gemm<<<gGemm, blk, 0, stream>>>(T1B, packW + (size_t)(s0 + 1) * 16384, acc, acc);
      k_prop<<<gNW, blk, 0, stream>>>(colp, csc_src, csc_w, P1, P2);
      k_combine<<<gVec, blk, 0, stream>>>(P2, inF, P2, T2B, NV4);
      k_gemm<<<gGemm, blk, 0, stream>>>(T2B, packW + (size_t)(s0 + 2) * 16384, acc, acc);
      k_post<<<gVec, blk, 0, stream>>>(acc, conv_b + (size_t)(r * 3 + l) * DIM,
                                       (l > 0 ? (const float*)h : (const float*)nullptr),
                                       h, HB, N_NODES);
      inF = h;
      inB = HB;
    }
  }

  // relation-attention fusion
  for (int r = 0; r < 3; r++) {
    k_convert<<<gVec, blk, 0, stream>>>(emb[r], T1B, NV4);
    k_gemm<<<gGemm, blk, 0, stream>>>(T1B, packW + (size_t)27 * 16384, (const float*)nullptr, acc);
    k_post<<<gVec, blk, 0, stream>>>(acc, gb1, (const float*)nullptr, P1, (bf16_t*)nullptr, N_NODES);
    k_matvec<<<gNW, blk, 0, stream>>>(P1, gW2, gb2, score + r, 3);
  }
  k_softmax_fuse<<<gNW, blk, 0, stream>>>(score, emb[0], emb[1], emb[2], P2);
  k_convert<<<gVec, blk, 0, stream>>>(P2, T1B, NV4);
  k_gemm<<<gGemm, blk, 0, stream>>>(T1B, packW + (size_t)28 * 16384, (const float*)nullptr, acc);
  k_post<<<gVec, blk, 0, stream>>>(acc, pb, (const float*)nullptr, P1, (bf16_t*)nullptr, N_NODES);
  k_convert<<<gVec, blk, 0, stream>>>(P1, T1B, NV4);
  k_gemm<<<gGemm, blk, 0, stream>>>(T1B, packW + (size_t)29 * 16384, (const float*)nullptr, acc);
  k_post<<<gVec, blk, 0, stream>>>(acc, cb1, (const float*)nullptr, P2, (bf16_t*)nullptr, N_NODES);
  k_matvec<<<gNW, blk, 0, stream>>>(P2, cW2, cb2, out, 1);
  for (int r = 0; r < 3; r++)
    k_matvec<<<gNW, blk, 0, stream>>>(emb[r], auxW + (size_t)r * DIM, auxb + r,
                                      out + (size_t)(1 + r) * N_NODES, 1);

  (void)in_sizes; (void)n_in; (void)out_size; (void)ws_size;
}
